// FFTFrequencyFilter_61375082660502
// MI455X (gfx1250) — matrix operand order repaired and checked by reading, not compiled, not run on hardware
//
#include <hip/hip_runtime.h>
#include <hip/hip_bf16.h>

// ---------------------------------------------------------------------------
// FFT high-pass filter == circular 11x11 conv (mask is real & centrosymmetric)
// followed by 1x1 channel conv (GEMM M=15488,N=2048,K=2048) + BN + ReLU + gate.
// GEMM: bf16 WMMA 16x16x32, 128x128 block tile, double-buffered async->LDS.
// ---------------------------------------------------------------------------

#define Bsz 128
#define Csz 2048
#define HW  121          // 11*11
#define Msz (Bsz * HW)   // 15488 = 121 * 128 (tile-exact)
#define NK  (Csz / 32)   // 64 K-steps

typedef __attribute__((ext_vector_type(16))) __bf16 v16bf;
typedef __attribute__((ext_vector_type(8)))  __bf16 v8bf;
typedef __attribute__((ext_vector_type(8)))  float  v8f;
typedef __attribute__((ext_vector_type(4)))  int    v4i;

#if defined(__has_builtin)
#if __has_builtin(__builtin_amdgcn_global_load_async_to_lds_b128) && \
    __has_builtin(__builtin_amdgcn_s_wait_asynccnt)
#define USE_ASYNC_LDS 1
#endif
#endif

// -------------------------------------------------------------------------
// Kernel 1: per-channel circular-conv taps  k_c = Re( ifft2( ifftshift(mask_c) ) )
// k[p,q] = (1/121) * sum_{u,v} Mu[u,v] * cos(2*pi*(u*p+v*q)/11)
// -------------------------------------------------------------------------
__global__ void fftfilt_taps(const float* __restrict__ thr, float* __restrict__ taps) {
    const int c = blockIdx.x;
    __shared__ float Mu[HW];
    const int t = threadIdx.x;
    const float cut = fminf(fmaxf(thr[c], 0.05f), 0.5f);
    if (t < HW) {
        const int us = t / 11, vs = t % 11;
        // ifftshift = roll by -(11//2) = -5  ->  Mu[u] = mask_shifted[(u+5) % 11]
        const int y  = (us + 5) % 11, xx = (vs + 5) % 11;
        const float dy = (float)(y - 5), dx = (float)(xx - 5);
        const float dn = sqrtf(dy * dy + dx * dx) / (sqrtf(50.0f) + 1e-6f);
        Mu[t] = 1.0f / (1.0f + __expf(-10.0f * (dn - cut)));
    }
    __syncthreads();
    if (t < HW) {
        const int p = t / 11, q = t % 11;
        float sum = 0.0f;
        #pragma unroll 1
        for (int u = 0; u < 11; ++u)
            for (int v = 0; v < 11; ++v) {
                const int ph = (u * p + v * q) % 11;
                sum += Mu[u * 11 + v] * __cosf(0.5711986642890533f * (float)ph); // 2*pi/11
            }
        taps[c * HW + t] = sum * (1.0f / 121.0f);
    }
}

// -------------------------------------------------------------------------
// Kernel 2: conv_w (f32 [o][c]) -> bf16 Wb [n=o][k=c]  (already B-friendly)
// -------------------------------------------------------------------------
__global__ void fftfilt_cvtw(const float* __restrict__ w, __bf16* __restrict__ wb, int n) {
    const int i = blockIdx.x * 256 + threadIdx.x;
    if (i < n) wb[i] = (__bf16)w[i];
}

// -------------------------------------------------------------------------
// Kernel 3: circular conv per (b, 64-channel tile) + pack bf16 into Xf[M][K]
// -------------------------------------------------------------------------
__global__ void fftfilt_filter(const float* __restrict__ x, const float* __restrict__ taps,
                               __bf16* __restrict__ Xf) {
    __shared__ float xs[64][123];
    __shared__ float ts[64][123];
    const int b  = blockIdx.y;
    const int c0 = blockIdx.x * 64;
    for (int i = threadIdx.x; i < 64 * HW; i += 256) {
        const int ci = i / HW, s = i - ci * HW;
        xs[ci][s] = x[((long long)(b * Csz + c0 + ci)) * HW + s];
        ts[ci][s] = taps[(c0 + ci) * HW + s];
    }
    __syncthreads();
    for (int i = threadIdx.x; i < 64 * HW; i += 256) {
        const int ci = i & 63, s = i >> 6;          // ci fastest -> coalesced bf16 stores
        const int h = s / 11, w = s - h * 11;
        float sum = 0.0f;
        #pragma unroll 1
        for (int p = 0; p < 11; ++p) {
            int hh = h - p; if (hh < 0) hh += 11;
            #pragma unroll
            for (int q = 0; q < 11; ++q) {
                int ww = w - q; if (ww < 0) ww += 11;
                sum += ts[ci][p * 11 + q] * xs[ci][hh * 11 + ww];
            }
        }
        Xf[((long long)(b * HW + s)) * Csz + (c0 + ci)] = (__bf16)sum;
    }
}

// -------------------------------------------------------------------------
// Kernel 4: bf16 WMMA GEMM  y[m,n] = sum_k Xf[m,k] * Wb[n,k]   (+ fused epilogue)
// Block tile 128x128, 8 waves as 2(M) x 4(N); wave tile 64x32 = 4x2 WMMA tiles.
// Double-buffered GLOBAL_LOAD_ASYNC_TO_LDS_B128 staging (ASYNCcnt pipelined).
// -------------------------------------------------------------------------
#define LDS_ROW 40   // 32 bf16 + 8 pad -> 80B rows, conflict-free b128 frag reads
#define TILE_BYTES (128 * LDS_ROW * 2)

__device__ __forceinline__ v16bf ld_frag(const __bf16* p0, const __bf16* p1) {
    v8bf a = *(const v8bf*)p0;
    v8bf b = *(const v8bf*)p1;
    v16bf r;
    #pragma unroll
    for (int i = 0; i < 8; ++i) { r[i] = a[i]; r[i + 8] = b[i]; }
    return r;
}

#if defined(USE_ASYNC_LDS)
typedef __attribute__((address_space(1))) v4i gv4i;   // global int4
typedef __attribute__((address_space(3))) v4i lv4i;   // LDS int4
__device__ __forceinline__ void async_ld16(const void* g, void* l) {
    __builtin_amdgcn_global_load_async_to_lds_b128((gv4i*)g, (lv4i*)l, 0, 0);
}
#endif

__global__ void __launch_bounds__(256)
fftfilt_gemm(const __bf16* __restrict__ Xf, const __bf16* __restrict__ Wb,
             const float* __restrict__ x,  const float* __restrict__ gate,
             const float* __restrict__ conv_b, const float* __restrict__ gamma,
             const float* __restrict__ beta, const float* __restrict__ mean,
             const float* __restrict__ var, float* __restrict__ out) {
    // One 40KB block: [bufA0|bufA1|bufB0|bufB1] during GEMM, reused by epilogue.
    __shared__ __align__(16) unsigned char smem[4 * TILE_BYTES];
    __bf16* ldsA = (__bf16*)smem;                       // [2][128][LDS_ROW]
    __bf16* ldsB = (__bf16*)(smem + 2 * TILE_BYTES);    // [2][128][LDS_ROW]

    const int tid  = threadIdx.x;
    const int lane = tid & 31;
    const int wave = tid >> 5;
    const int half = lane >> 4;     // 0: lanes 0-15, 1: lanes 16-31
    const int lr   = lane & 15;
    const int wm   = (wave >> 2) * 64;  // wave M offset in block tile
    const int wn   = (wave & 3) * 32;   // wave N offset in block tile
    const int m0   = blockIdx.y * 128;  // block M origin (M divisible by 128)
    const int n0   = blockIdx.x * 128;  // block N origin

    // Per-thread staging coordinates: 2 chunks of 16B per matrix per K-step.
    const int row0 = tid >> 2,            col0 = (tid & 3) * 8;
    const int row1 = (tid + 256) >> 2,    col1 = col0;

    v8f acc[4][2];
    #pragma unroll
    for (int mt = 0; mt < 4; ++mt)
        #pragma unroll
        for (int nt = 0; nt < 2; ++nt)
            acc[mt][nt] = (v8f){0.f, 0.f, 0.f, 0.f, 0.f, 0.f, 0.f, 0.f};

#if defined(USE_ASYNC_LDS)
    // ---- async double-buffered pipeline --------------------------------
    {   // prologue: issue tile kt=0 into buffer 0
        const int k0 = 0;
        async_ld16(&Xf[((long long)(m0 + row0)) * Csz + k0 + col0], &ldsA[row0 * LDS_ROW + col0]);
        async_ld16(&Xf[((long long)(m0 + row1)) * Csz + k0 + col1], &ldsA[row1 * LDS_ROW + col1]);
        async_ld16(&Wb[((long long)(n0 + row0)) * Csz + k0 + col0], &ldsB[row0 * LDS_ROW + col0]);
        async_ld16(&Wb[((long long)(n0 + row1)) * Csz + k0 + col1], &ldsB[row1 * LDS_ROW + col1]);
    }
    for (int kt = 0; kt < NK; ++kt) {
        const int cur = kt & 1;
        if (kt + 1 < NK) {      // issue next tile into the other buffer
            const int nb = (cur ^ 1) * 128 * LDS_ROW;
            const int k0 = (kt + 1) * 32;
            async_ld16(&Xf[((long long)(m0 + row0)) * Csz + k0 + col0], &ldsA[nb + row0 * LDS_ROW + col0]);
            async_ld16(&Xf[((long long)(m0 + row1)) * Csz + k0 + col1], &ldsA[nb + row1 * LDS_ROW + col1]);
            async_ld16(&Wb[((long long)(n0 + row0)) * Csz + k0 + col0], &ldsB[nb + row0 * LDS_ROW + col0]);
            async_ld16(&Wb[((long long)(n0 + row1)) * Csz + k0 + col1], &ldsB[nb + row1 * LDS_ROW + col1]);
            __builtin_amdgcn_s_wait_asynccnt(4);  // in-order: current tile landed
        } else {
            __builtin_amdgcn_s_wait_asynccnt(0);
        }
        __syncthreads();                          // all waves' current tile visible

        const __bf16* tA = ldsA + cur * 128 * LDS_ROW;
        const __bf16* tB = ldsB + cur * 128 * LDS_ROW;
        v16bf bfrag[2], afrag[4];
        #pragma unroll
        for (int nt = 0; nt < 2; ++nt) {
            const __bf16* r = &tB[(wn + nt * 16 + lr) * LDS_ROW];
            bfrag[nt] = ld_frag(r + half * 16, r + half * 16 + 8);
        }
        #pragma unroll
        for (int mt = 0; mt < 4; ++mt) {
            const __bf16* r = &tA[(wm + mt * 16 + lr) * LDS_ROW];
            afrag[mt] = ld_frag(r + half * 16, r + half * 16 + 8);
        }
        #pragma unroll
        for (int mt = 0; mt < 4; ++mt)
            #pragma unroll
            for (int nt = 0; nt < 2; ++nt)
                acc[mt][nt] = __builtin_amdgcn_wmma_f32_16x16x32_bf16(
                    false, afrag[mt], false, bfrag[nt], (short)0, acc[mt][nt],
                    false, false);
        __syncthreads();   // WAR: next iteration's async issue overwrites this buffer
    }
#else
    // ---- fallback: synchronous single-buffered staging -----------------
    for (int kt = 0; kt < NK; ++kt) {
        const int k0 = kt * 32;
        *(v8bf*)&ldsA[row0 * LDS_ROW + col0] = *(const v8bf*)&Xf[((long long)(m0 + row0)) * Csz + k0 + col0];
        *(v8bf*)&ldsA[row1 * LDS_ROW + col1] = *(const v8bf*)&Xf[((long long)(m0 + row1)) * Csz + k0 + col1];
        *(v8bf*)&ldsB[row0 * LDS_ROW + col0] = *(const v8bf*)&Wb[((long long)(n0 + row0)) * Csz + k0 + col0];
        *(v8bf*)&ldsB[row1 * LDS_ROW + col1] = *(const v8bf*)&Wb[((long long)(n0 + row1)) * Csz + k0 + col1];
        __syncthreads();
        v16bf bfrag[2], afrag[4];
        #pragma unroll
        for (int nt = 0; nt < 2; ++nt) {
            const __bf16* r = &ldsB[(wn + nt * 16 + lr) * LDS_ROW];
            bfrag[nt] = ld_frag(r + half * 16, r + half * 16 + 8);
        }
        #pragma unroll
        for (int mt = 0; mt < 4; ++mt) {
            const __bf16* r = &ldsA[(wm + mt * 16 + lr) * LDS_ROW];
            afrag[mt] = ld_frag(r + half * 16, r + half * 16 + 8);
        }
        #pragma unroll
        for (int mt = 0; mt < 4; ++mt)
            #pragma unroll
            for (int nt = 0; nt < 2; ++nt)
                acc[mt][nt] = __builtin_amdgcn_wmma_f32_16x16x32_bf16(
                    false, afrag[mt], false, bfrag[nt], (short)0, acc[mt][nt],
                    false, false);
        __syncthreads();
    }
#endif

    // ---- fused epilogue, coalesced via per-wave LDS transpose ----------
    // out = x + g * relu((y + conv_b - mean)*gamma*rsqrt(var+eps) + beta)
    __syncthreads();                       // tile buffers are dead; reuse LDS
    float* epw = (float*)smem + wave * (64 * 17);   // per-wave 64x17 f32 (4.25KB)
    const float g = fminf(fmaxf(gate[0], 0.0f), 1.0f);
    #pragma unroll
    for (int nt = 0; nt < 2; ++nt) {
        // scatter accumulators (D layout: row = mt*16 + half*8 + r, col = lr)
        #pragma unroll
        for (int mt = 0; mt < 4; ++mt)
            #pragma unroll
            for (int r = 0; r < 8; ++r)
                epw[(mt * 16 + half * 8 + r) * 17 + lr] = acc[mt][nt][r];
        // wave-private region: DS ops are in-order within a wave, no barrier.
        #pragma unroll 1
        for (int j = 0; j < 16; ++j) {
            const int n = n0 + wn + nt * 16 + j;          // uniform -> SALU loads
            const float sc  = gamma[n] * rsqrtf(var[n] + 1e-5f);
            const float add = (conv_b[n] - mean[n]) * sc + beta[n];
            #pragma unroll
            for (int mm2 = 0; mm2 < 2; ++mm2) {
                const int m_local = mm2 * 32 + lane;      // lanes -> consecutive m
                const float yv = epw[m_local * 17 + j];
                const int m = m0 + wm + m_local;          // m = b*121 + s
                const int b = m / HW;
                const int s = m - b * HW;
                const long long idx = ((long long)(b * Csz + n)) * HW + s;
                out[idx] = x[idx] + g * fmaxf(yv * sc + add, 0.0f);
            }
        }
    }
}

// ---------------------------------------------------------------------------
extern "C" void kernel_launch(void* const* d_in, const int* in_sizes, int n_in,
                              void* d_out, int out_size, void* d_ws, size_t ws_size,
                              hipStream_t stream) {
    const float* x      = (const float*)d_in[0];
    const float* thr    = (const float*)d_in[1];
    const float* gate   = (const float*)d_in[2];
    const float* conv_w = (const float*)d_in[3];
    const float* conv_b = (const float*)d_in[4];
    const float* gamma  = (const float*)d_in[5];
    const float* beta   = (const float*)d_in[6];
    const float* mean   = (const float*)d_in[7];
    const float* var    = (const float*)d_in[8];
    float* out = (float*)d_out;

    char* ws = (char*)d_ws;
    float*  taps = (float*)ws;                                   // 2048*121*4  ~ 0.95 MB
    __bf16* Wb   = (__bf16*)(ws + (1 << 20));                    // 2048*2048*2 =  8 MB
    __bf16* Xf   = (__bf16*)(ws + (1 << 20) + (8 << 20));        // 15488*2048*2 ~ 60.5 MB

    fftfilt_taps<<<Csz, 128, 0, stream>>>(thr, taps);
    fftfilt_cvtw<<<(Csz * Csz + 255) / 256, 256, 0, stream>>>(conv_w, Wb, Csz * Csz);
    fftfilt_filter<<<dim3(Csz / 64, Bsz), 256, 0, stream>>>(x, taps, Xf);
    fftfilt_gemm<<<dim3(Csz / 128, Msz / 128), 256, 0, stream>>>(
        Xf, Wb, x, gate, conv_b, gamma, beta, mean, var, out);
}
